// GanCraftGenerator_89163521065324
// MI455X (gfx1250) — compile-verified, matrix operand order checked by reading, not run on hardware
//
#include <hip/hip_runtime.h>
#include <math.h>

typedef _Float16 f16;
typedef __attribute__((ext_vector_type(16))) _Float16 v16h;
typedef __attribute__((ext_vector_type(8)))  float    v8f;

#define BM 128
#define BN 128
#define BK 32
#define BKP 40   // padded LDS row in halves (avoid bank conflicts)

static inline int KP(int k) { return (k + 31) & ~31; }

__device__ __forceinline__ float lrelu(float v) { return v > 0.f ? v : 0.2f * v; }

// --- CDNA5 async global->LDS staging (ASYNCcnt) with sync fallback ----------
#if __has_builtin(__builtin_amdgcn_global_load_async_to_lds_b128)
#define ASYNC_LDS 1
typedef int v4i_ __attribute__((vector_size(16)));
typedef v4i_ __attribute__((address_space(1)))* gas_v4i;
typedef v4i_ __attribute__((address_space(3)))* las_v4i;
__device__ __forceinline__ void cp_async16(f16* lds_dst, const f16* gsrc) {
  __builtin_amdgcn_global_load_async_to_lds_b128(
      (gas_v4i)gsrc, (las_v4i)lds_dst, 0, 0);
}
#else
#define ASYNC_LDS 0
__device__ __forceinline__ void cp_async16(f16* lds_dst, const f16* gsrc) {
  *(uint4*)lds_dst = *(const uint4*)gsrc;
}
#endif

__device__ __forceinline__ void wait_async0() {
#if __has_builtin(__builtin_amdgcn_s_wait_asynccnt)
  __builtin_amdgcn_s_wait_asynccnt(0);
#elif ASYNC_LDS
  asm volatile("s_wait_asynccnt 0x0" ::: "memory");
#endif
}

// ---------------------------------------------------------------------------
// Core WMMA GEMM:  O[M,N] (+= epilogue)  =  A[M,K] * Bw[N,K]^T
//   CONV=0: A is row-major f16 with leading dim lda (K multiple of 32)
//   CONV=1: A gathered on the fly from NHWC image Xc (implicit im2col),
//           k = tap*Cin + ci, tap = ky*KW + kx
// Block = 256 threads = 8 waves (wave32), tile 128x128.
// Wave tile 32x64 -> 2x4 v_wmma_f32_16x16x32_f16 fragments (8 WMMA / K-step),
// double-buffered LDS staged with async-to-LDS copies, f32 accumulation.
// Epilogue: +bias[n], +fcm[n,label[m]], +res[m,n], act(0 none/1 leaky/2 tanh),
// f16 store into O at column offset coff with leading dim ldc.
// ---------------------------------------------------------------------------
template<int CONV>
__global__ __launch_bounds__(256)
void k_wmma_gemm(const f16* __restrict__ A, int lda,
                 const f16* __restrict__ Xc, int Cin, int IH, int IW,
                 int KH, int KW, int stride, int pad, int OW,
                 const f16* __restrict__ Bw, int ldb, int N, int K,
                 f16* __restrict__ O, int ldc, int coff,
                 const float* __restrict__ bias,
                 const float* __restrict__ fcm, const int* __restrict__ labels,
                 const f16* __restrict__ res, int act)
{
  __shared__ __align__(16) f16 lA[2][BM][BKP];
  __shared__ __align__(16) f16 lB[2][BN][BKP];
  const int tid   = threadIdx.x;
  const int m_blk = blockIdx.x * BM;
  const int n_blk = blockIdx.y * BN;
  const int lane  = tid & 31;
  const int wid   = tid >> 5;
  const int wm    = (wid & 3) * 32;    // wave m offset (4 waves in m)
  const int wn    = (wid >> 2) * 64;   // wave n offset (2 waves in n)
  const int half  = lane >> 4;
  const int l16   = lane & 15;

  v8f acc[2][4] = {};
  const int Kloop = (K + BK - 1) & ~(BK - 1);
  const int nstep = Kloop / BK;

  auto fill = [&](int buf, int k0) {
    if constexpr (CONV) {
      // scalar gather fill: 128x32 halves, 16 per thread
      #pragma unroll
      for (int i = 0; i < 16; ++i) {
        int e   = tid + i * 256;
        int row = e >> 5;
        int kk  = (e & 31) + k0;
        f16 v = (f16)0.f;
        if (kk < K) {
          int tap = kk / Cin;
          int ci  = kk - tap * Cin;
          int om  = m_blk + row;
          int oy  = om / OW;
          int ox  = om - oy * OW;
          int ky  = tap / KW;
          int kx  = tap - ky * KW;
          int iy  = oy * stride - pad + ky;
          int ix  = ox * stride - pad + kx;
          if (iy >= 0 && iy < IH && ix >= 0 && ix < IW)
            v = Xc[(size_t)(iy * IW + ix) * Cin + ci];
        }
        lA[buf][row][e & 31] = v;
      }
    } else {
      // async b128 copies: 512 x 16B, 2 per thread
      #pragma unroll
      for (int i = 0; i < 2; ++i) {
        int vi  = tid + i * 256;
        int row = vi >> 2;
        int ch  = vi & 3;
        cp_async16(&lA[buf][row][ch * 8],
                   A + (size_t)(m_blk + row) * lda + k0 + ch * 8);
      }
    }
    // B tile: 128 rows x 32 halves, async where valid, zero elsewhere
    #pragma unroll
    for (int i = 0; i < 2; ++i) {
      int vi  = tid + i * 256;
      int row = vi >> 2;
      int ch  = vi & 3;
      f16* dst = &lB[buf][row][ch * 8];
      if (n_blk + row < N) {
        cp_async16(dst, Bw + (size_t)(n_blk + row) * ldb + k0 + ch * 8);
      } else {
        uint4 z4; z4.x = z4.y = z4.z = z4.w = 0u;
        *(uint4*)dst = z4;
      }
    }
  };

  fill(0, 0);
  wait_async0();
  __syncthreads();

  for (int it = 0; it < nstep; ++it) {
    int cur = it & 1;
    if (it + 1 < nstep) fill(cur ^ 1, (it + 1) * BK);  // overlap with WMMAs

    // fragment loads per documented 16-bit A 16x32 layout (mirrored for B):
    // lane half h holds K chunks [8h,8h+8) and [16+8h,24+8h)
    v16h af[2], bf[4];
    #pragma unroll
    for (int mi = 0; mi < 2; ++mi) {
      const f16* rp = &lA[cur][wm + mi * 16 + l16][0];
      union { uint4 u[2]; v16h h; } c;
      c.u[0] = *(const uint4*)(rp + 8 * half);
      c.u[1] = *(const uint4*)(rp + 16 + 8 * half);
      af[mi] = c.h;
    }
    #pragma unroll
    for (int ni = 0; ni < 4; ++ni) {
      const f16* rp = &lB[cur][wn + ni * 16 + l16][0];
      union { uint4 u[2]; v16h h; } c;
      c.u[0] = *(const uint4*)(rp + 8 * half);
      c.u[1] = *(const uint4*)(rp + 16 + 8 * half);
      bf[ni] = c.h;
    }
    #pragma unroll
    for (int mi = 0; mi < 2; ++mi)
      #pragma unroll
      for (int ni = 0; ni < 4; ++ni)
        acc[mi][ni] = __builtin_amdgcn_wmma_f32_16x16x32_f16(
            false, af[mi], false, bf[ni], (short)0, acc[mi][ni], false, false);

    wait_async0();
    __syncthreads();
  }

  // epilogue — C/D layout: VGPR r, lane l -> m = r + 8*(l/16), n = l%16
  #pragma unroll
  for (int mi = 0; mi < 2; ++mi) {
    #pragma unroll
    for (int ni = 0; ni < 4; ++ni) {
      int n = n_blk + wn + ni * 16 + l16;
      if (n >= N) continue;
      #pragma unroll
      for (int r = 0; r < 8; ++r) {
        int m = m_blk + wm + mi * 16 + half * 8 + r;
        float v = acc[mi][ni][r];
        if (bias) v += bias[n];
        if (fcm)  v += fcm[n * 9 + labels[m]];
        if (res)  v += (float)res[(size_t)m * ldc + coff + n];
        if (act == 1)      v = lrelu(v);
        else if (act == 2) v = tanhf(v);
        O[(size_t)m * ldc + coff + n] = (f16)v;
      }
    }
  }
}

// ---------------------------------------------------------------------------
// Weight pack: f32 [Cout][Cin][KH][KW] -> f16 [Cout][Kpad], k = tap*Cin+ci,
// zero-padded to Kpad, optional per-input-channel alpha scaling (ModLinear).
// ---------------------------------------------------------------------------
__global__ void k_wcvt(const float* __restrict__ src, f16* __restrict__ dst,
                       int Cout, int Cin, int KH, int KW, int Kpad,
                       const float* __restrict__ alpha) {
  int i = blockIdx.x * 256 + threadIdx.x;
  if (i >= Cout * Kpad) return;
  int co = i / Kpad, k = i - co * Kpad;
  int K = Cin * KH * KW;
  f16 v = (f16)0.f;
  if (k < K) {
    int tap = k / Cin, ci = k - tap * Cin;
    int ky = tap / KW, kx = tap - ky * KW;
    float w = src[(((size_t)co * Cin + ci) * KH + ky) * KW + kx];
    if (alpha) w *= alpha[ci];
    v = (f16)w;
  }
  dst[i] = v;
}

// ModLinear vectors for B=1: alpha = z@wa^T+ba, folded bias = b + z@wb^T+bb
__global__ void k_modvec(const float* __restrict__ z, const float* __restrict__ wa,
                         const float* __restrict__ ba, const float* __restrict__ wb,
                         const float* __restrict__ bb, const float* __restrict__ b,
                         float* __restrict__ alpha, float* __restrict__ biasf) {
  int j = blockIdx.x * 256 + threadIdx.x;
  if (j >= 256) return;
  float sa = 0.f, sb = 0.f;
  for (int k = 0; k < 256; ++k) {
    float zk = z[k];
    sa += zk * wa[j * 256 + k];
    sb += zk * wb[j * 256 + k];
  }
  alpha[j] = sa + ba[j];
  biasf[j] = sb + bb[j] + b[j];
}

__global__ void k_cnn_a(const float* __restrict__ z, const float* __restrict__ w,
                        const float* __restrict__ b, float* __restrict__ a) {
  int j = blockIdx.x * 256 + threadIdx.x;
  if (j >= 1024) return;
  float s = 0.f;
  for (int k = 0; k < 256; ++k) s += z[k] * w[j * 256 + k];
  a[j] = s + b[j];
}

// split NCHW f32 input into NHWC f16 hf(1ch) + seg(9ch)
__global__ void k_split(const float* __restrict__ hf_seg, f16* __restrict__ hf,
                        f16* __restrict__ seg) {
  int p = blockIdx.x * 256 + threadIdx.x;
  if (p >= 65536) return;
  hf[p] = (f16)hf_seg[p];
  for (int c = 0; c < 9; ++c)
    seg[p * 9 + c] = (f16)hf_seg[(size_t)(1 + c) * 65536 + p];
}

// GroupNorm(32 groups)+ReLU on an NHWC f16 channel slice; one block per group
__global__ __launch_bounds__(256)
void k_gn_relu(const f16* __restrict__ src, int sld, int scoff,
               f16* __restrict__ dst, int dld,
               const float* __restrict__ g, const float* __restrict__ b,
               int C, int P) {
  __shared__ float ss[256], sq[256];
  int grp = blockIdx.x;
  int gsz = C >> 5;
  int c0 = grp * gsz;
  int total = P * gsz;
  float s = 0.f, q = 0.f;
  for (int i = threadIdx.x; i < total; i += 256) {
    int p = i / gsz;
    int c = c0 + (i - p * gsz);
    float v = (float)src[(size_t)p * sld + scoff + c];
    s += v; q += v * v;
  }
  ss[threadIdx.x] = s; sq[threadIdx.x] = q;
  __syncthreads();
  for (int o = 128; o > 0; o >>= 1) {
    if ((int)threadIdx.x < o) {
      ss[threadIdx.x] += ss[threadIdx.x + o];
      sq[threadIdx.x] += sq[threadIdx.x + o];
    }
    __syncthreads();
  }
  float inv  = 1.f / (float)total;
  float mu   = ss[0] * inv;
  float var  = sq[0] * inv - mu * mu;
  float rstd = rsqrtf(var + 1e-5f);
  for (int i = threadIdx.x; i < total; i += 256) {
    int p = i / gsz;
    int c = c0 + (i - p * gsz);
    float v = (float)src[(size_t)p * sld + scoff + c];
    v = (v - mu) * rstd * g[c] + b[c];
    dst[(size_t)p * dld + c] = (f16)fmaxf(v, 0.f);
  }
}

__global__ void k_avgpool(const f16* __restrict__ in, f16* __restrict__ out,
                          int Ho, int Wo, int C) {
  int i = blockIdx.x * 256 + threadIdx.x;
  if (i >= Ho * Wo * C) return;
  int c = i % C;
  int p = i / C;
  int y = p / Wo, x = p - y * Wo;
  int Wi = Wo * 2;
  const f16* r0 = in + ((size_t)(2 * y) * Wi + 2 * x) * C + c;
  const f16* r1 = in + ((size_t)(2 * y + 1) * Wi + 2 * x) * C + c;
  float v = (float)r0[0] + (float)r0[C] + (float)r1[0] + (float)r1[C];
  out[(size_t)p * C + c] = (f16)(0.25f * v);
}

// ConvTranspose2d(k=4,s=2,p=1): each output pixel sums exactly 2x2 taps.
// torch weight layout [Cin][Cout][4][4]. NHWC f16 activations, f32 accum.
__global__ void k_convt(const f16* __restrict__ in, const float* __restrict__ w,
                        const float* __restrict__ bias, f16* __restrict__ out,
                        int IH, int IW, int Cin, int Cout) {
  int OH = IH * 2, OW = IW * 2;
  int idx = blockIdx.x * 256 + threadIdx.x;
  if (idx >= OH * OW * Cout) return;
  int co  = idx % Cout;
  int pix = idx / Cout;
  int oy = pix / OW, ox = pix - oy * OW;
  float acc = bias[co];
  int ky0 = (oy + 1) & 1, kx0 = (ox + 1) & 1;
  for (int ty = 0; ty < 2; ++ty) {
    int ky = ky0 + 2 * ty;
    int iy = (oy + 1 - ky) >> 1;
    if (iy < 0 || iy >= IH) continue;
    for (int tx = 0; tx < 2; ++tx) {
      int kx = kx0 + 2 * tx;
      int ix = (ox + 1 - kx) >> 1;
      if (ix < 0 || ix >= IW) continue;
      const f16*   ip = in + (size_t)(iy * IW + ix) * Cin;
      const float* wp = w + (size_t)co * 16 + (ky << 2) + kx;
      for (int ci = 0; ci < Cin; ++ci)
        acc += (float)ip[ci] * wp[(size_t)ci * Cout * 16];
    }
  }
  out[(size_t)pix * Cout + co] = (f16)acc;
}

// bilinear sample 63 feature channels + NeRF sin/cos encoding -> X0[M,96] f16
__global__ void k_sample(const f16* __restrict__ feats, const float* __restrict__ cxy,
                         const float* __restrict__ cz, f16* __restrict__ X0) {
  int p = blockIdx.x * 256 + threadIdx.x;
  if (p >= 147456) return;
  float px = cxy[2 * p] * 255.f;
  float py = cxy[2 * p + 1] * 255.f;
  float fx = floorf(px), fy = floorf(py);
  float wx = px - fx, wy = py - fy;
  int x0 = (int)fx; x0 = x0 < 0 ? 0 : (x0 > 255 ? 255 : x0);
  int y0 = (int)fy; y0 = y0 < 0 ? 0 : (y0 > 255 ? 255 : y0);
  int x1 = x0 + 1 > 255 ? 255 : x0 + 1;
  int y1 = y0 + 1 > 255 ? 255 : y0 + 1;
  const f16* f00 = feats + ((size_t)y0 * 256 + x0) * 64;
  const f16* f01 = feats + ((size_t)y0 * 256 + x1) * 64;
  const f16* f10 = feats + ((size_t)y1 * 256 + x0) * 64;
  const f16* f11 = feats + ((size_t)y1 * 256 + x1) * 64;
  float w00 = (1.f - wx) * (1.f - wy), w01 = wx * (1.f - wy);
  float w10 = (1.f - wx) * wy, w11 = wx * wy;
  f16* o = X0 + (size_t)p * 96;
  for (int c = 0; c < 63; ++c)
    o[c] = (f16)(w00 * (float)f00[c] + w01 * (float)f01[c] +
                 w10 * (float)f10[c] + w11 * (float)f11[c]);
  float zz = cz[p];
  for (int j = 0; j < 8; ++j) {
    float fr = (float)(1 << j);
    o[63 + j] = (f16)sinf(zz * fr);
    o[71 + j] = (f16)cosf(zz * fr);
  }
  for (int j = 79; j < 96; ++j) o[j] = (f16)0.f;
}

__global__ void k_sigma(const f16* __restrict__ h4, const float* __restrict__ sw,
                        const float* __restrict__ sb, float* __restrict__ sig) {
  int p = blockIdx.x * 256 + threadIdx.x;
  if (p >= 147456) return;
  float acc = sb[0];
  const f16* hp = h4 + (size_t)p * 256;
  for (int i = 0; i < 256; ++i) acc += (float)hp[i] * sw[i];
  sig[p] = acc;
}

// volume compositing: block per pixel, thread per color channel
__global__ void k_volume(const float* __restrict__ sig, const float* __restrict__ deltas,
                         const f16* __restrict__ cfeat, f16* __restrict__ fmap) {
  int pix = blockIdx.x;
  int c = threadIdx.x;   // 64 threads
  float T = 1.f, acc = 0.f;
  for (int s = 0; s < 16; ++s) {
    int m = pix * 16 + s;
    float a = 1.f - expf(-fmaxf(sig[m], 0.f) * deltas[m]);
    acc += a * T * (float)cfeat[(size_t)m * 64 + c];
    T *= (1.f - a + 1e-10f);
  }
  fmap[(size_t)pix * 64 + c] = (f16)acc;
}

__global__ void k_modact(f16* __restrict__ y, const float* __restrict__ a0,
                         const float* __restrict__ a1) {
  int i = blockIdx.x * 256 + threadIdx.x;
  if (i >= 9216 * 256) return;
  int c = i & 255;
  float v = (float)y[i];
  v = v * (a0[c] + 1.f) + a1[c];
  y[i] = (f16)lrelu(v);
}

// final 1x1 conv 256->3 (leaky on input), CHW f32 output
__global__ void k_final(const f16* __restrict__ y3, const float* __restrict__ w,
                        const float* __restrict__ b, float* __restrict__ out) {
  int pix = blockIdx.x * 256 + threadIdx.x;
  if (pix >= 9216) return;
  float acc0 = b[0], acc1 = b[1], acc2 = b[2];
  const f16* yp = y3 + (size_t)pix * 256;
  for (int k = 0; k < 256; ++k) {
    float v = lrelu((float)yp[k]);
    acc0 += v * w[k];
    acc1 += v * w[256 + k];
    acc2 += v * w[512 + k];
  }
  out[pix]          = acc0;
  out[9216 + pix]   = acc1;
  out[18432 + pix]  = acc2;
}

// ---------------------------------------------------------------------------
// host-side launch helpers
// ---------------------------------------------------------------------------
static void gemm(hipStream_t s, const f16* A, int lda, const f16* Bw, int ldb,
                 f16* O, int ldc, int coff, int M, int N, int K,
                 const float* bias, int act,
                 const f16* res = nullptr,
                 const float* fcm = nullptr, const int* labels = nullptr) {
  dim3 g(M / BM, (N + BN - 1) / BN);
  k_wmma_gemm<0><<<g, 256, 0, s>>>(A, lda, nullptr, 0, 0, 0, 0, 0, 0, 0, 0,
                                   Bw, ldb, N, K, O, ldc, coff,
                                   bias, fcm, labels, res, act);
}

static void convg(hipStream_t s, const f16* X, int Cin, int IH, int IW,
                  int KH, int KW, int stride, int pad, int OH, int OW,
                  const f16* Bw, int ldb, f16* O, int ldc, int coff,
                  int N, const float* bias, int act, const f16* res = nullptr) {
  int M = OH * OW;
  dim3 g(M / BM, (N + BN - 1) / BN);
  k_wmma_gemm<1><<<g, 256, 0, s>>>(nullptr, 0, X, Cin, IH, IW, KH, KW, stride,
                                   pad, OW, Bw, ldb, N, Cin * KH * KW,
                                   O, ldc, coff, bias, nullptr, nullptr, res, act);
}

static void wcvt(hipStream_t s, const float* src, f16* dst, int Cout,
                 int Cin, int KH, int KW, const float* alpha = nullptr) {
  int Kpad = KP(Cin * KH * KW);
  int tot = Cout * Kpad;
  k_wcvt<<<(tot + 255) / 256, 256, 0, s>>>(src, dst, Cout, Cin, KH, KW, Kpad, alpha);
}

struct RBW { f16 *w1, *w2, *w3, *wd; };

static void run_rb(hipStream_t s, const f16* x, f16* big, f16* tbuf,
                   int Cin, int Cout, int H, int W,
                   const float* g1, const float* b1, const float* g2, const float* b2,
                   const float* g3, const float* b3, const float* g4, const float* b4,
                   const RBW& wp) {
  int P = H * W;
  int C2 = Cout / 2, C4 = Cout / 4;
  k_gn_relu<<<32, 256, 0, s>>>(x, Cin, 0, tbuf, Cin, g1, b1, Cin, P);
  convg(s, tbuf, Cin, H, W, 3, 3, 1, 1, H, W, wp.w1, KP(Cin * 9), big, Cout, 0, C2, nullptr, 0);
  k_gn_relu<<<32, 256, 0, s>>>(big, Cout, 0, tbuf, C2, g2, b2, C2, P);
  convg(s, tbuf, C2, H, W, 3, 3, 1, 1, H, W, wp.w2, KP(C2 * 9), big, Cout, C2, C4, nullptr, 0);
  k_gn_relu<<<32, 256, 0, s>>>(big, Cout, C2, tbuf, C4, g3, b3, C4, P);
  convg(s, tbuf, C4, H, W, 3, 3, 1, 1, H, W, wp.w3, KP(C4 * 9), big, Cout, C2 + C4, C4, nullptr, 0);
  k_gn_relu<<<32, 256, 0, s>>>(x, Cin, 0, tbuf, Cin, g4, b4, Cin, P);
  gemm(s, tbuf, Cin, wp.wd, KP(Cin), big, Cout, 0, P, Cout, KP(Cin), nullptr, 0, big);
}

// ---------------------------------------------------------------------------
// d_in order (insertion order of setup_inputs, params flattened depth-first):
// 0 hf_seg 1 z 2 coords_xy 3 coords_z 4 deltas 5 labels
// enc: 6 hf_w 7 hf_b 8 seg_w 9 seg_b 10 g1 11 b1
//   rb2: 12..23 (g1,b1,w1,g2,b2,w2,g3,b3,w3,g4,b4,wd)  rb3: 24..35  rb4: 36..47
//   48 d5_w 49 d5_b 50 d6_w 51 d6_b 52 d7_w 53 d7_b
// mlp: 54 fc1_w 55 fc1_b 56 fcm_w  fc2:57..62(w,b,wa,ba,wb,bb) fc3:63 fc4:69
//   75 sig_w 76 sig_b fc5:77 fc6:83  89 c_w 90 c_b
// cnn: 91 fc_w 92 fc_b 93 c1_w 94 c1_b 95 c2a_w 96 c2a_b 97 c2b_w
//   98 c3a_w 99 c3a_b 100 c3b_w 101 c4a_w 102 c4a_b 103 c4b_w 104 c4b_b
//   105 c4_w 106 c4_b
// ---------------------------------------------------------------------------
extern "C" void kernel_launch(void* const* d_in, const int* in_sizes, int n_in,
                              void* d_out, int out_size, void* d_ws, size_t ws_size,
                              hipStream_t stream) {
  (void)in_sizes; (void)n_in; (void)out_size; (void)ws_size;
  hipStream_t s = stream;
  auto F = [&](int i) { return (const float*)d_in[i]; };
  const int* labels = (const int*)d_in[5];
  const float* z = F(1);

  // deterministic bump allocator over workspace (~190 MB total)
  char* wsb = (char*)d_ws;
  size_t off = 0;
  auto alloc  = [&](size_t bytes) -> char* {
    char* p = wsb + off; off += (bytes + 255) & ~(size_t)255; return p;
  };
  auto allocH = [&](size_t n) { return (f16*)alloc(n * 2); };
  auto allocF = [&](size_t n) { return (float*)alloc(n * 4); };

  // persistent vectors + packed weights
  float* alpha5 = allocF(5 * 256);
  float* biasf5 = allocF(5 * 256);
  float* acnn   = allocF(1024);
  float* sig    = allocF(147456);

  f16* fc1p = allocH(256 * 96);
  f16* wmod[5]; for (int l = 0; l < 5; ++l) wmod[l] = allocH(256 * 256);
  f16* cwp  = allocH(64 * 256);
  f16* c1p  = allocH(256 * 64);
  f16* c2ap = allocH((size_t)256 * 2304);
  f16* c2bp = allocH((size_t)256 * 2304);
  f16* c3ap = allocH((size_t)256 * 2304);
  f16* c3bp = allocH((size_t)256 * 2304);
  f16* c4ap = allocH(256 * 256);
  f16* c4bp = allocH(256 * 256);
  f16* hfp  = allocH(32 * 64);
  f16* segp = allocH(32 * 448);
  RBW rb2w{allocH(64 * 576),   allocH(32 * 576),   allocH(32 * 288),   allocH(128 * 64)};
  RBW rb3w{allocH(128 * 1152), allocH(64 * 1152),  allocH(64 * 576),   allocH(256 * 128)};
  RBW rb4w{allocH((size_t)256 * 2304), allocH((size_t)128 * 2304),
           allocH(128 * 1152), allocH(512 * 256)};
  f16* d7p = allocH(63 * 32);

  // big ping-pong regions
  f16* R1 = allocH((size_t)147456 * 256);   // h1/h3/h5, later CNN buffers
  f16* R2 = allocH((size_t)147456 * 256);   // encoder buffers, h2/h4/h6
  f16* X0 = allocH((size_t)147456 * 96);    // MLP input, later cfeat+fmap

  // encoder buffers live inside R2 (dead before fc2 writes R2)
  char* eb = (char*)R2; size_t eoff = 0;
  auto ealloc = [&](size_t n) -> f16* {
    f16* p = (f16*)(eb + eoff); eoff += (n * 2 + 255) & ~(size_t)255; return p;
  };
  f16* hf_x  = ealloc(65536);
  f16* seg_x = ealloc((size_t)65536 * 9);
  f16* e1    = ealloc((size_t)16384 * 64);
  f16* x2    = ealloc((size_t)16384 * 64);
  f16* tbuf  = ealloc((size_t)16384 * 128);
  f16* big2  = ealloc((size_t)16384 * 128);
  f16* pool2 = ealloc((size_t)4096 * 128);
  f16* big3  = ealloc((size_t)4096 * 256);
  f16* big4  = ealloc((size_t)4096 * 512);
  f16* d5o   = ealloc((size_t)16384 * 128);
  f16* d6o   = ealloc((size_t)65536 * 32);
  f16* feats = ealloc((size_t)65536 * 64);

  f16* cfeat = X0;                                  // X0 dead after fc1
  f16* fmap  = X0 + (size_t)147456 * 64;
  f16* y0 = R1;                                     // R1 dead after fc6
  f16* yt = R1 + (size_t)9216 * 256;
  f16* y1 = R1 + (size_t)2 * 9216 * 256;

  // --- phase 0: z-dependent vectors + weight packing (f32 -> f16, K padded)
  int mlb[5] = {57, 63, 69, 77, 83};
  for (int l = 0; l < 5; ++l)
    k_modvec<<<1, 256, 0, s>>>(z, F(mlb[l] + 2), F(mlb[l] + 3), F(mlb[l] + 4),
                               F(mlb[l] + 5), F(mlb[l] + 1),
                               alpha5 + l * 256, biasf5 + l * 256);
  k_cnn_a<<<4, 256, 0, s>>>(z, F(91), F(92), acnn);

  wcvt(s, F(54), fc1p, 256, 79, 1, 1);
  for (int l = 0; l < 5; ++l)
    wcvt(s, F(mlb[l]), wmod[l], 256, 256, 1, 1, alpha5 + l * 256);  // fold alpha
  wcvt(s, F(89),  cwp,  64, 256, 1, 1);
  wcvt(s, F(93),  c1p,  256, 64, 1, 1);
  wcvt(s, F(95),  c2ap, 256, 256, 3, 3);
  wcvt(s, F(97),  c2bp, 256, 256, 3, 3);
  wcvt(s, F(98),  c3ap, 256, 256, 3, 3);
  wcvt(s, F(100), c3bp, 256, 256, 3, 3);
  wcvt(s, F(101), c4ap, 256, 256, 1, 1);
  wcvt(s, F(103), c4bp, 256, 256, 1, 1);
  wcvt(s, F(6),  hfp,  32, 1, 7, 7);
  wcvt(s, F(8),  segp, 32, 9, 7, 7);
  wcvt(s, F(14), rb2w.w1, 64, 64, 3, 3);   wcvt(s, F(17), rb2w.w2, 32, 64, 3, 3);
  wcvt(s, F(20), rb2w.w3, 32, 32, 3, 3);   wcvt(s, F(23), rb2w.wd, 128, 64, 1, 1);
  wcvt(s, F(26), rb3w.w1, 128, 128, 3, 3); wcvt(s, F(29), rb3w.w2, 64, 128, 3, 3);
  wcvt(s, F(32), rb3w.w3, 64, 64, 3, 3);   wcvt(s, F(35), rb3w.wd, 256, 128, 1, 1);
  wcvt(s, F(38), rb4w.w1, 256, 256, 3, 3); wcvt(s, F(41), rb4w.w2, 128, 256, 3, 3);
  wcvt(s, F(44), rb4w.w3, 128, 128, 3, 3); wcvt(s, F(47), rb4w.wd, 512, 256, 1, 1);
  wcvt(s, F(52), d7p, 63, 32, 1, 1);

  // --- phase 1: local encoder (NHWC f16, WMMA implicit-GEMM convs)
  k_split<<<256, 256, 0, s>>>(F(0), hf_x, seg_x);
  convg(s, hf_x, 1, 256, 256, 7, 7, 2, 3, 128, 128, hfp, 64, e1, 64, 0, 32, F(7), 0);
  convg(s, seg_x, 9, 256, 256, 7, 7, 2, 3, 128, 128, segp, 448, e1, 64, 32, 32, F(9), 0);
  k_gn_relu<<<32, 256, 0, s>>>(e1, 64, 0, x2, 64, F(10), F(11), 64, 16384);
  run_rb(s, x2, big2, tbuf, 64, 128, 128, 128,
         F(12), F(13), F(15), F(16), F(18), F(19), F(21), F(22), rb2w);
  k_avgpool<<<(4096 * 128 + 255) / 256, 256, 0, s>>>(big2, pool2, 64, 64, 128);
  run_rb(s, pool2, big3, tbuf, 128, 256, 64, 64,
         F(24), F(25), F(27), F(28), F(30), F(31), F(33), F(34), rb3w);
  run_rb(s, big3, big4, tbuf, 256, 512, 64, 64,
         F(36), F(37), F(39), F(40), F(42), F(43), F(45), F(46), rb4w);
  k_convt<<<(16384 * 128 + 255) / 256, 256, 0, s>>>(big4, F(48), F(49), d5o, 64, 64, 512, 128);
  k_convt<<<(65536 * 32 + 255) / 256, 256, 0, s>>>(d5o, F(50), F(51), d6o, 128, 128, 128, 32);
  gemm(s, d6o, 32, d7p, 32, feats, 64, 0, 65536, 63, 32, F(53), 2);  // tanh

  // --- phase 2: sampling + positional encoding
  k_sample<<<576, 256, 0, s>>>(feats, F(2), F(3), X0);

  // --- phase 3: render MLP (dominant: 107 GFLOP of WMMA GEMMs)
  gemm(s, X0, 96, fc1p, 96, R1, 256, 0, 147456, 256, 96, F(55), 1, nullptr, F(56), labels);
  gemm(s, R1, 256, wmod[0], 256, R2, 256, 0, 147456, 256, 256, biasf5 + 0,    1);
  gemm(s, R2, 256, wmod[1], 256, R1, 256, 0, 147456, 256, 256, biasf5 + 256,  1);
  gemm(s, R1, 256, wmod[2], 256, R2, 256, 0, 147456, 256, 256, biasf5 + 512,  1);
  k_sigma<<<576, 256, 0, s>>>(R2, F(75), F(76), sig);
  gemm(s, R2, 256, wmod[3], 256, R1, 256, 0, 147456, 256, 256, biasf5 + 768,  1);
  gemm(s, R1, 256, wmod[4], 256, R2, 256, 0, 147456, 256, 256, biasf5 + 1024, 1);
  gemm(s, R2, 256, cwp, 256, cfeat, 64, 0, 147456, 64, 256, F(90), 0);

  // --- phase 4: volume compositing
  k_volume<<<9216, 64, 0, s>>>(sig, F(4), cfeat, fmap);

  // --- phase 5: render CNN (NHWC f16, WMMA)
  gemm(s, fmap, 64, c1p, 64, y0, 256, 0, 9216, 256, 64, F(94), 1);
  convg(s, y0, 256, 96, 96, 3, 3, 1, 1, 96, 96, c2ap, 2304, yt, 256, 0, 256, F(96), 1);
  convg(s, yt, 256, 96, 96, 3, 3, 1, 1, 96, 96, c2bp, 2304, y1, 256, 0, 256, nullptr, 0, y0);
  k_modact<<<9216, 256, 0, s>>>(y1, acnn + 0, acnn + 256);
  convg(s, y1, 256, 96, 96, 3, 3, 1, 1, 96, 96, c3ap, 2304, yt, 256, 0, 256, F(99), 1);
  convg(s, yt, 256, 96, 96, 3, 3, 1, 1, 96, 96, c3bp, 2304, y0, 256, 0, 256, nullptr, 0, y1);
  k_modact<<<9216, 256, 0, s>>>(y0, acnn + 512, acnn + 768);
  gemm(s, y0, 256, c4ap, 256, yt, 256, 0, 9216, 256, 256, F(102), 1);
  gemm(s, yt, 256, c4bp, 256, y1, 256, 0, 9216, 256, 256, F(104), 0, y0);
  k_final<<<36, 256, 0, s>>>(y1, F(105), F(106), (float*)d_out);
}
